// BlockRowColMultiHeadAttention_28269474742394
// MI455X (gfx1250) — matrix operand order repaired and checked by reading, not compiled, not run on hardware
//
#include <hip/hip_runtime.h>

typedef _Float16 v16h __attribute__((ext_vector_type(16)));
typedef _Float16 v8h  __attribute__((ext_vector_type(8)));
typedef float    v8f  __attribute__((ext_vector_type(8)));

#define WMMA_F16(a,b,c) __builtin_amdgcn_wmma_f32_16x16x32_f16( \
    false, (a), false, (b), (short)0, (c), false, false)

// ---- geometry ----
// DIM=64, NUM_HEADS=4, HEAD_DIM=16, WINDOW=64, SCALE=0.25
// row pass: 2*512 rows * 8 windows = 8192 blocks; col pass same.

// LDS strides (elements). 72 halves = 144B and 40 halves = 80B are odd multiples
// of 16B -> conflict-free across 16 rows; 66 floats for the f32 score buffer.
#define LDX 72
#define LDQ 40
#define LDS_S 66

// LDS layout (bytes), time-multiplexed:
//  region A: Xh (stage1-2, 9216) -> Sf (stage3-4, 67584) -> Oh (stage5-6, 9216)
//  region B: Qp+Kp (stage1-3, 20480+20480) -> Pp (stage4-5, 36864)
//  region E: Vt (stage2-5, 9216)
#define OFF_A   0
#define OFF_B   67584
#define OFF_KP  (67584 + 20480)
#define OFF_VT  (67584 + 40960)
#define SMEM_BYTES (67584 + 40960 + 9216)   // 117760 B

// A fragment: 16x32 f16, row-major in LDS. lane<16: row=lane, K=[k0+0..7]+[k0+16..23];
// lane>=16: row=lane-16, K=[k0+8..15]+[k0+24..31]  (ISA 7.12.2, 16-bit A 16x32)
__device__ __forceinline__ v16h frag_a(const _Float16* base, int ld, int k0, int lane) {
    int m = lane & 15, sel = lane >> 4;
    const _Float16* p = base + m * ld + k0 + 8 * sel;
    union { v16h v; v8h h[2]; } u;
    u.h[0] = *(const v8h*)(p);
    u.h[1] = *(const v8h*)(p + 16);
    return u.v;
}

// B fragment: 32x16 f16, sourced from an N-major (B^T) image.
// lane<16: col n=lane, K=[k0+0..7]+[k0+16..23] ; lane>=16: col n=lane-16, K=[k0+8..15]+[k0+24..31]
__device__ __forceinline__ v16h frag_bt(const _Float16* baseT, int ld, int k0, int lane) {
    int n = lane & 15, sel = lane >> 4;
    const _Float16* p = baseT + n * ld + k0 + 8 * sel;
    union { v16h v; v8h h[2]; } u;
    u.h[0] = *(const v8h*)(p);
    u.h[1] = *(const v8h*)(p + 16);
    return u.v;
}

// Convert fp32 weights to f16, transposed to N-major for frag_bt.
// ws[0..12287]     = Wqkv^T (192 x 64)
// ws[12288..16383] = Wout^T ( 64 x 64)
__global__ __launch_bounds__(256) void conv_weights_kernel(
    const float* __restrict__ w_qkv, const float* __restrict__ w_out,
    _Float16* __restrict__ ws) {
    int tid = blockIdx.x * 256 + threadIdx.x;
    if (tid < 12288) {
        int c = tid >> 6, k = tid & 63;           // Wqkv is (64 x 192) row-major
        ws[tid] = (_Float16)w_qkv[k * 192 + c];
    } else if (tid < 16384) {
        int t2 = tid - 12288;
        int c = t2 >> 6, k = t2 & 63;             // Wout is (64 x 64) row-major
        ws[tid] = (_Float16)w_out[k * 64 + c];
    }
}

__global__ __launch_bounds__(256) void attn_block_kernel(
    const float* x, float* out,
    const _Float16* __restrict__ wqkvT, const _Float16* __restrict__ woutT,
    const float* __restrict__ bqkv, const float* __restrict__ bout,
    int col_mode) {
    __shared__ __align__(16) char smem[SMEM_BYTES];
    _Float16* Xh = (_Float16*)(smem + OFF_A);
    float*    Sf = (float*)   (smem + OFF_A);
    _Float16* Oh = (_Float16*)(smem + OFF_A);
    _Float16* Qp = (_Float16*)(smem + OFF_B);
    _Float16* Kp = (_Float16*)(smem + OFF_KP);
    _Float16* Pp = (_Float16*)(smem + OFF_B);
    _Float16* Vt = (_Float16*)(smem + OFF_VT);

    const int tid  = threadIdx.x;
    const int lane = tid & 31;
    // wave id is uniform within a wave: force it into an SGPR so all tile loops
    // and the QKV scatter use scalar branches (EXEC stays all-1s across WMMA).
    const int wv   = __builtin_amdgcn_readfirstlane(tid >> 5);
    const int n_   = lane & 15;
    const int sel  = lane >> 4;

    // block -> global base + token stride
    size_t base, tstride;
    {
        int bb = blockIdx.x;
        if (!col_mode) {
            int r = bb >> 3, jw = bb & 7;                       // (b*L+i) row, window along j
            base = ((size_t)r * 512 + (size_t)jw * 64) * 64;
            tstride = 64;
        } else {
            int rc = bb >> 3, b = rc >> 9, j = rc & 511, iw = bb & 7;  // window along i
            base = (((size_t)b * 512 + (size_t)iw * 64) * 512 + (size_t)j) * 64;
            tstride = (size_t)512 * 64;
        }
    }

    // keep the (tiny, hot) weight images resident
    __builtin_prefetch(wqkvT + (size_t)(tid & 63) * 256, 0, 0);

    // ---- stage 1: global X (64x64 f32) -> Xh f16; zero K-pad of Qp/Kp ----
    {
        int t  = tid >> 2;
        int c0 = (tid & 3) * 16;
        const float* g = x + base + (size_t)t * tstride + c0;
        _Float16* d = Xh + t * LDX + c0;
        #pragma unroll
        for (int q = 0; q < 4; ++q) {
            union { float4 f; float a[4]; } u;
            u.f = *(const float4*)(g + q * 4);
            d[q * 4 + 0] = (_Float16)u.a[0];
            d[q * 4 + 1] = (_Float16)u.a[1];
            d[q * 4 + 2] = (_Float16)u.a[2];
            d[q * 4 + 3] = (_Float16)u.a[3];
        }
        uint4 z = {0u, 0u, 0u, 0u};
        *(uint4*)(Qp + tid * LDQ + 16) = z;     // halves 16..23
        *(uint4*)(Qp + tid * LDQ + 24) = z;     // halves 24..31
        *(uint4*)(Kp + tid * LDQ + 16) = z;
        *(uint4*)(Kp + tid * LDQ + 24) = z;
    }
    __syncthreads();

    // ---- stage 2: QKV = Xh @ Wqkv + b ; scatter into Qp/Kp/Vt (f16) ----
    #pragma unroll
    for (int it = 0; it < 6; ++it) {                      // 48 tiles / 8 waves
        int tile = wv + it * 8;
        int mi = tile / 12, cj = tile % 12;
        v8f acc = {};
        #pragma unroll
        for (int ks = 0; ks < 2; ++ks) {
            v16h a = frag_a(Xh + mi * 16 * LDX, LDX, ks * 32, lane);
            v16h b = frag_bt(wqkvT + (cj * 16) * 64, 64, ks * 32, lane);
            acc = WMMA_F16(a, b, acc);
        }
        int c = cj * 16 + n_;
        float bias = bqkv[c];
        if (cj < 4) {                                      // Q
            int h = c >> 4, dch = c & 15;
            _Float16* dst = Qp + (h * 64) * LDQ + dch;
            #pragma unroll
            for (int i = 0; i < 8; ++i) {
                int t = mi * 16 + i + 8 * sel;
                dst[t * LDQ] = (_Float16)(acc[i] + bias);
            }
        } else if (cj < 8) {                               // K
            int c2 = c - 64; int h = c2 >> 4, dch = c2 & 15;
            _Float16* dst = Kp + (h * 64) * LDQ + dch;
            #pragma unroll
            for (int i = 0; i < 8; ++i) {
                int t = mi * 16 + i + 8 * sel;
                dst[t * LDQ] = (_Float16)(acc[i] + bias);
            }
        } else {                                           // V, stored transposed (d-major)
            int c2 = c - 128; int h = c2 >> 4, dch = c2 & 15;
            _Float16* dst = Vt + (h * 16 + dch) * LDX;
            #pragma unroll
            for (int i = 0; i < 8; ++i) {
                int t = mi * 16 + i + 8 * sel;
                dst[t] = (_Float16)(acc[i] + bias);
            }
        }
    }
    __syncthreads();

    // ---- stage 3: S = scale * Q K^T (per head, K=16 zero-padded to 32) ----
    #pragma unroll
    for (int it = 0; it < 8; ++it) {                      // 64 tiles / 8 waves
        int tile = wv + it * 8;
        int h = tile >> 4, rem = tile & 15, ti = rem >> 2, tj = rem & 3;
        v16h a = frag_a(Qp + (h * 64 + ti * 16) * LDQ, LDQ, 0, lane);
        v16h b = frag_bt(Kp + (h * 64 + tj * 16) * LDQ, LDQ, 0, lane);
        v8f acc = {};
        acc = WMMA_F16(a, b, acc);
        float* dst = Sf + (h * 64 + ti * 16 + 8 * sel) * LDS_S + tj * 16 + n_;
        #pragma unroll
        for (int i = 0; i < 8; ++i) dst[i * LDS_S] = acc[i] * 0.25f;
    }
    __syncthreads();

    // ---- stage 4: row softmax (f32) -> Pp f16 ----
    {
        int h = tid >> 6, row = tid & 63;
        float* s = Sf + (h * 64 + row) * LDS_S;
        float mx = -3.0e38f;
        #pragma unroll 8
        for (int j = 0; j < 64; ++j) mx = fmaxf(mx, s[j]);
        float sum = 0.0f;
        #pragma unroll 8
        for (int j = 0; j < 64; ++j) { float e = __expf(s[j] - mx); sum += e; s[j] = e; }
        float inv = __frcp_rn(sum);
        _Float16* p = Pp + (h * 64 + row) * LDX;
        #pragma unroll 8
        for (int j = 0; j < 64; ++j) p[j] = (_Float16)(s[j] * inv);
    }
    __syncthreads();

    // ---- stage 5: O = P @ V  (per head, K=64) ----
    #pragma unroll
    for (int it = 0; it < 2; ++it) {                      // 16 tiles / 8 waves
        int tile = wv + it * 8;
        int h = tile >> 2, ti = tile & 3;
        v8f acc = {};
        #pragma unroll
        for (int ks = 0; ks < 2; ++ks) {
            v16h a = frag_a(Pp + (h * 64 + ti * 16) * LDX, LDX, ks * 32, lane);
            v16h b = frag_bt(Vt + (h * 16) * LDX, LDX, ks * 32, lane);
            acc = WMMA_F16(a, b, acc);
        }
        #pragma unroll
        for (int i = 0; i < 8; ++i) {
            int t = ti * 16 + i + 8 * sel;
            Oh[t * LDX + h * 16 + n_] = (_Float16)acc[i];
        }
    }
    __syncthreads();

    // ---- stage 6: out = Oh @ Wout + b_out -> global f32 ----
    #pragma unroll
    for (int it = 0; it < 2; ++it) {                      // 16 tiles / 8 waves
        int tile = wv + it * 8;
        int mi = tile >> 2, cj = tile & 3;
        v8f acc = {};
        #pragma unroll
        for (int ks = 0; ks < 2; ++ks) {
            v16h a = frag_a(Oh + mi * 16 * LDX, LDX, ks * 32, lane);
            v16h b = frag_bt(woutT + (cj * 16) * 64, 64, ks * 32, lane);
            acc = WMMA_F16(a, b, acc);
        }
        int c = cj * 16 + n_;
        float bias = bout[c];
        float* g = out + base + c;
        #pragma unroll
        for (int i = 0; i < 8; ++i) {
            int t = mi * 16 + i + 8 * sel;
            g[(size_t)t * tstride] = acc[i] + bias;
        }
    }
}

extern "C" void kernel_launch(void* const* d_in, const int* in_sizes, int n_in,
                              void* d_out, int out_size, void* d_ws, size_t ws_size,
                              hipStream_t stream) {
    const float* pair  = (const float*)d_in[0];
    const float* w_qkv = (const float*)d_in[1];
    const float* b_qkv = (const float*)d_in[2];
    const float* w_out = (const float*)d_in[3];
    const float* b_out = (const float*)d_in[4];
    float* out = (float*)d_out;

    _Float16* wh = (_Float16*)d_ws;               // 16384 halves = 32 KB
    conv_weights_kernel<<<64, 256, 0, stream>>>(w_qkv, w_out, wh);

    const _Float16* wqkvT = wh;
    const _Float16* woutT = wh + 12288;

    // row pass: pair -> out
    attn_block_kernel<<<8192, 256, 0, stream>>>(pair, out, wqkvT, woutT, b_qkv, b_out, 0);
    // col pass: in-place on out (each block reads exactly the elements it writes)
    attn_block_kernel<<<8192, 256, 0, stream>>>(out, out, wqkvT, woutT, b_qkv, b_out, 1);
}
